// NodeDeepOnet_58514634441396
// MI455X (gfx1250) — compile-verified, matrix operand order checked by reading, not run on hardware
//
#include <hip/hip_runtime.h>
#include <hip/hip_bf16.h>

// ---------------------------------------------------------------------------
// NodeDeepOnet on MI455X (gfx1250, wave32, WMMA).
// All GEMMs run through v_wmma_f32_16x16x32_bf16 (f32 accumulate).
// Compute-bound (~67 GFLOP, ~100MB live data -> L2 resident), so the bf16
// matrix pipe is the right path. B-operand tiles are staged into LDS via
// GLOBAL_LOAD_ASYNC_TO_LDS (ASYNCcnt) when the toolchain exposes it.
// ---------------------------------------------------------------------------

typedef __bf16 bf16_t;
typedef __attribute__((ext_vector_type(16))) __bf16 v16bf;
typedef __attribute__((ext_vector_type(8)))  __bf16 v8bf;
typedef __attribute__((ext_vector_type(8)))  float  v8f;
typedef int v4i_t __attribute__((__vector_size__(16)));
typedef __attribute__((address_space(1))) v4i_t* gv4i_p;  // global int4*
typedef __attribute__((address_space(3))) v4i_t* lv4i_p;  // LDS int4*

static constexpr int kB   = 8;
static constexpr int kQ   = 1024;
static constexpr int kN   = 2048;   // NMAX
static constexpr int kH   = 384;
static constexpr int kOut = 3;
static constexpr int kHH  = kH * kH;           // 147456
static constexpr int kKN  = kB * kN;           // 16384 known rows
static constexpr int kQT  = kB * kQ;           // 8192 query rows

#if __has_builtin(__builtin_amdgcn_global_load_async_to_lds_b128)
#define HAS_ASYNC_LDS 1
#endif

__device__ __forceinline__ void stage16(const bf16_t* g, bf16_t* l) {
#ifdef HAS_ASYNC_LDS
  __builtin_amdgcn_global_load_async_to_lds_b128((gv4i_p)(g), (lv4i_p)(l), 0, 0);
#else
  *(v8bf*)l = *(const v8bf*)g;
#endif
}

__device__ __forceinline__ void stage_wait() {
#ifdef HAS_ASYNC_LDS
#if __has_builtin(__builtin_amdgcn_s_wait_asynccnt)
  __builtin_amdgcn_s_wait_asynccnt(0);
#else
  asm volatile("s_wait_asynccnt 0x0" ::: "memory");
#endif
#endif
}

__device__ __forceinline__ v8f wmma_bf16(v16bf a, v16bf b, v8f c) {
  // (neg_a, A, neg_b, B, c_mod, C, reuse_a, reuse_b)
  return __builtin_amdgcn_wmma_f32_16x16x32_bf16(false, a, false, b,
                                                 (short)0, c, false, false);
}

// Load one 16x32 bf16 A/B fragment from global memory. ISA layout: per-lane
// row = row0 + (lane&15); lanes 0-15 khalf=0, lanes 16-31 khalf=8;
// VGPR0..3 = K[khalf..khalf+7], VGPR4..7 = K[16+khalf..23+khalf].
template <bool AF32>
__device__ __forceinline__ v16bf load_frag(const void* __restrict__ src,
                                           int row, int ld, int kk, int lane) {
  const int r = row + (lane & 15);
  const int khalf = (lane >> 4) << 3;
  union { v16bf v; v8bf h[2]; } u;
  if constexpr (AF32) {
    const float* p = (const float*)src + (size_t)r * ld + kk + khalf;
    v8f lo = *(const v8f*)p;
    v8f hi = *(const v8f*)(p + 16);
#pragma unroll
    for (int i = 0; i < 8; ++i) { u.v[i] = (bf16_t)lo[i]; u.v[8 + i] = (bf16_t)hi[i]; }
  } else {
    const bf16_t* p = (const bf16_t*)src + (size_t)r * ld + kk + khalf;
    u.h[0] = *(const v8bf*)p;
    u.h[1] = *(const v8bf*)(p + 16);
  }
  return u.v;
}

// Same fragment layout, reading a [128 x 32] bf16 tile held in LDS.
__device__ __forceinline__ v16bf load_frag_lds(const bf16_t* __restrict__ sW,
                                               int rowLocal, int lane) {
  const int r = rowLocal + (lane & 15);
  const int khalf = (lane >> 4) << 3;
  union { v16bf v; v8bf h[2]; } u;
  const bf16_t* p = sW + r * 32 + khalf;
  u.h[0] = *(const v8bf*)p;
  u.h[1] = *(const v8bf*)(p + 16);
  return u.v;
}

// Y[M,N] = act(scale * (X[M,K] @ W[N,K]^T) + bias)
// Block: 256 threads = 8 waves as 2(M) x 4(N); wave tile 64x32 (4x2 WMMA).
// Block tile: 128(M) x 128(N). W tile [128 x 32] staged in LDS per K-step
// (async copy when available). All call sites divide exactly (EXEC all-1s).
// OMODE: 0 = bf16 row-major, 1 = bf16 transposed per-batch (for V^T),
//        2 = f32 row-major (attention scores).
template <bool AF32, int OMODE, bool RELU>
__global__ __launch_bounds__(256) void gemm_wmma_k(
    const void* __restrict__ A, const bf16_t* __restrict__ W,
    const float* __restrict__ bias, void* __restrict__ Out,
    int M, int N, int K, int ldA, int ldW, int ldOut,
    float scale, int transRows) {
  __shared__ __align__(16) bf16_t sW[128 * 32];  // 8 KB

  const int tid  = threadIdx.x;
  const int lane = tid & 31;
  const int wave = tid >> 5;
  const int m0   = blockIdx.y * 128 + (wave >> 2) * 64;
  const int nblk = blockIdx.x * 128;
  const int nw   = (wave & 3) * 32;  // wave's N offset within the block tile

  // Cooperative staging: 8KB tile = 512 x 16B chunks; 2 chunks per thread.
  const int srow = tid >> 2;             // 0..63
  const int soff = (tid & 3) * 8;        // element offset within 32-wide row
  const bf16_t* gstage = W + (size_t)(nblk + srow) * ldW + soff;
  bf16_t* lstage = sW + srow * 32 + soff;

  v8f acc[4][2] = {};
  for (int kk = 0; kk < K; kk += 32) {
    // stage W[nblk..nblk+127, kk..kk+31] -> LDS
    stage16(gstage + kk, lstage);
    stage16(gstage + (size_t)64 * ldW + kk, lstage + 64 * 32);
    stage_wait();
    __syncthreads();

    v16bf a0 = load_frag<AF32>(A, m0 +  0, ldA, kk, lane);
    v16bf a1 = load_frag<AF32>(A, m0 + 16, ldA, kk, lane);
    v16bf a2 = load_frag<AF32>(A, m0 + 32, ldA, kk, lane);
    v16bf a3 = load_frag<AF32>(A, m0 + 48, ldA, kk, lane);
    if (kk + 64 < K) {  // prefetch A two K-steps ahead (global_prefetch_b8)
      const char* pf = AF32
          ? (const char*)((const float*)A + (size_t)(m0 + (lane & 15)) * ldA + kk + 64)
          : (const char*)((const bf16_t*)A + (size_t)(m0 + (lane & 15)) * ldA + kk + 64);
      __builtin_prefetch(pf, 0, 1);
      __builtin_prefetch(pf + (AF32 ? 4u : 2u) * 32u * (unsigned)ldA, 0, 1);
    }
    v16bf b0 = load_frag_lds(sW, nw +  0, lane);
    v16bf b1 = load_frag_lds(sW, nw + 16, lane);

    acc[0][0] = wmma_bf16(a0, b0, acc[0][0]);
    acc[0][1] = wmma_bf16(a0, b1, acc[0][1]);
    acc[1][0] = wmma_bf16(a1, b0, acc[1][0]);
    acc[1][1] = wmma_bf16(a1, b1, acc[1][1]);
    acc[2][0] = wmma_bf16(a2, b0, acc[2][0]);
    acc[2][1] = wmma_bf16(a2, b1, acc[2][1]);
    acc[3][0] = wmma_bf16(a3, b0, acc[3][0]);
    acc[3][1] = wmma_bf16(a3, b1, acc[3][1]);
    __syncthreads();  // all reads done before next K-step overwrites sW
  }

  // D layout: VGPR i holds row m0'+i (lanes 0-15) / m0'+8+i (lanes 16-31),
  // column n0' + (lane&15).
  const int rbase = (lane >> 4) * 8;
  const int col_l = lane & 15;
#pragma unroll
  for (int mi = 0; mi < 4; ++mi)
#pragma unroll
    for (int ni = 0; ni < 2; ++ni) {
      const int colc = nblk + nw + ni * 16 + col_l;
      const float bv = bias ? bias[colc] : 0.f;
#pragma unroll
      for (int i = 0; i < 8; ++i) {
        const int row = m0 + mi * 16 + rbase + i;
        float v = acc[mi][ni][i] * scale + bv;
        if (RELU) v = fmaxf(v, 0.f);
        if constexpr (OMODE == 0) {
          ((bf16_t*)Out)[(size_t)row * ldOut + colc] = (bf16_t)v;
        } else if constexpr (OMODE == 1) {
          const int bb = row / transRows, rr = row - bb * transRows;
          ((bf16_t*)Out)[(size_t)bb * N * transRows + (size_t)colc * transRows + rr] = (bf16_t)v;
        } else {
          ((float*)Out)[(size_t)row * ldOut + colc] = v;
        }
      }
    }
}

// Sinusoidal 3D positional embedding -> H=384 bf16 channels.
// h -> coord j = h/128, r = h%128, f = r%64; r<64: sin, else cos.
__global__ void posemb_k(const float* __restrict__ pos,
                         bf16_t* __restrict__ out) {
  const int r = blockIdx.x;
  const int h = threadIdx.x;
  const int j = h >> 7, rr = h & 127, f = rr & 63, iscos = rr >> 6;
  const float p = pos[r * 3 + j] * 100.f;
  const float freq = __expf(-logf(10000.f) * (float)f / 64.f);
  const float ang = p * freq;
  const float v = iscos ? __cosf(ang) : __sinf(ang);
  out[(size_t)r * kH + h] = (bf16_t)v;
}

// In-place masked softmax over one score row (NMAX=2048 f32), one block/row.
__global__ __launch_bounds__(256) void softmax_k(float* __restrict__ sc,
                                                 const int* __restrict__ num_nodes,
                                                 int b) {
  __shared__ float red[256];
  const int tid = threadIdx.x;
  const int nn = num_nodes[b];
  float* base = sc + (size_t)blockIdx.x * kN;

  float vals[8];
  float mx = -INFINITY;
#pragma unroll
  for (int j = 0; j < 8; ++j) {
    const int i = tid + j * 256;
    const float x = (i < nn) ? base[i] : -INFINITY;
    vals[j] = x;
    mx = fmaxf(mx, x);
  }
  red[tid] = mx; __syncthreads();
  for (int s = 128; s > 0; s >>= 1) {
    if (tid < s) red[tid] = fmaxf(red[tid], red[tid + s]);
    __syncthreads();
  }
  mx = red[0]; __syncthreads();

  float sum = 0.f;
#pragma unroll
  for (int j = 0; j < 8; ++j) { vals[j] = __expf(vals[j] - mx); sum += vals[j]; }
  red[tid] = sum; __syncthreads();
  for (int s = 128; s > 0; s >>= 1) {
    if (tid < s) red[tid] += red[tid + s];
    __syncthreads();
  }
  const float inv = 1.f / red[0];
#pragma unroll
  for (int j = 0; j < 8; ++j) base[tid + j * 256] = vals[j] * inv;
}

// Final head: [QT,384] bf16 @ ow2[3,384]^T -> f32 [QT,3]. Tiny; plain VALU.
__global__ void head_k(const bf16_t* __restrict__ x, const float* __restrict__ w,
                       float* __restrict__ out, int rows) {
  const int t = blockIdx.x * blockDim.x + threadIdx.x;
  if (t >= rows * kOut) return;
  const int r = t / kOut, o = t - r * kOut;
  const bf16_t* xr = x + (size_t)r * kH;
  const float* wr = w + (size_t)o * kH;
  float s = 0.f;
  for (int h = 0; h < kH; ++h) s += (float)xr[h] * wr[h];
  out[t] = s;
}

struct WP12 { const float* p[12]; };

__global__ void cvt_weights_k(WP12 wp, bf16_t* __restrict__ dst) {
  const int idx = blockIdx.x * blockDim.x + threadIdx.x;
  if (idx >= 12 * kHH) return;
  const int m = idx / kHH, o = idx - m * kHH;
  dst[idx] = (bf16_t)wp.p[m][o];
}

extern "C" void kernel_launch(void* const* d_in, const int* in_sizes, int n_in,
                              void* d_out, int out_size, void* d_ws, size_t ws_size,
                              hipStream_t stream) {
  (void)in_sizes; (void)n_in; (void)out_size; (void)ws_size;

  const float* known_pos   = (const float*)d_in[0];
  const float* test_points = (const float*)d_in[1];
  const int*   num_nodes   = (const int*)d_in[2];
  const float* bb0 = (const float*)d_in[4];
  const float* bb1 = (const float*)d_in[6];
  const float* bb2 = (const float*)d_in[8];
  const float* tb0 = (const float*)d_in[10];
  const float* tb1 = (const float*)d_in[12];
  const float* tb2 = (const float*)d_in[14];
  const float* in_proj_w = (const float*)d_in[15];
  const float* in_proj_b = (const float*)d_in[16];
  const float* out_b = (const float*)d_in[18];
  const float* ob0 = (const float*)d_in[20];
  const float* ob1 = (const float*)d_in[22];
  const float* ow2 = (const float*)d_in[23];
  const float* bq = in_proj_b;
  const float* bk = in_proj_b + kH;
  const float* bv = in_proj_b + 2 * kH;

  // ---- workspace layout (~113 MB) ----
  char* ws = (char*)d_ws;
  size_t off = 0;
  auto alloc = [&](size_t bytes) -> void* {
    void* p = ws + off;
    off += (bytes + 255) & ~(size_t)255;
    return p;
  };
  bf16_t* Wb    = (bf16_t*)alloc((size_t)12 * kHH * 2);   // 12 bf16 weight mats
  bf16_t* kn_e  = (bf16_t*)alloc((size_t)kKN * kH * 2);
  bf16_t* tp_e  = (bf16_t*)alloc((size_t)kQT * kH * 2);
  bf16_t* tA    = (bf16_t*)alloc((size_t)kKN * kH * 2);
  bf16_t* fbuf  = (bf16_t*)alloc((size_t)kKN * kH * 2);
  bf16_t* cbuf  = (bf16_t*)alloc((size_t)kQT * kH * 2);
  bf16_t* qbuf  = (bf16_t*)alloc((size_t)kQT * kH * 2);
  bf16_t* kbuf  = (bf16_t*)alloc((size_t)kKN * kH * 2);
  bf16_t* vTbuf = (bf16_t*)alloc((size_t)kKN * kH * 2);   // per batch [H, NMAX]
  float*  sc    = (float*)alloc((size_t)kQ * kN * 4);     // per-batch scores
  bf16_t* attn  = (bf16_t*)alloc((size_t)kQT * kH * 2);
  bf16_t* t1    = (bf16_t*)alloc((size_t)kQT * kH * 2);
  bf16_t* t2    = (bf16_t*)alloc((size_t)kQT * kH * 2);

  // ---- weights -> bf16 ----
  WP12 wp;
  wp.p[0]  = (const float*)d_in[3];   // bw0
  wp.p[1]  = (const float*)d_in[5];   // bw1
  wp.p[2]  = (const float*)d_in[7];   // bw2
  wp.p[3]  = (const float*)d_in[9];   // tw0
  wp.p[4]  = (const float*)d_in[11];  // tw1
  wp.p[5]  = (const float*)d_in[13];  // tw2
  wp.p[6]  = in_proj_w;               // wq
  wp.p[7]  = in_proj_w + kHH;         // wk
  wp.p[8]  = in_proj_w + 2 * kHH;     // wv
  wp.p[9]  = (const float*)d_in[17];  // out_w
  wp.p[10] = (const float*)d_in[19];  // ow0
  wp.p[11] = (const float*)d_in[21];  // ow1
  cvt_weights_k<<<(12 * kHH + 255) / 256, 256, 0, stream>>>(wp, Wb);

  // ---- positional embeddings ----
  posemb_k<<<kKN, kH, 0, stream>>>(known_pos, kn_e);
  posemb_k<<<kQT, kH, 0, stream>>>(test_points, tp_e);

  const dim3 gKN(kH / 128, kKN / 128);  // [16384 x 384]
  const dim3 gQT(kH / 128, kQT / 128);  // [ 8192 x 384]

  // ---- brunch MLP over known nodes: f ----
  gemm_wmma_k<false, 0, true ><<<gKN, 256, 0, stream>>>(kn_e, Wb + 0 * kHH, bb0, tA,   kKN, kH, kH, kH, kH, kH, 1.f, 0);
  gemm_wmma_k<false, 0, true ><<<gKN, 256, 0, stream>>>(tA,   Wb + 1 * kHH, bb1, kn_e, kKN, kH, kH, kH, kH, kH, 1.f, 0);
  gemm_wmma_k<false, 0, false><<<gKN, 256, 0, stream>>>(kn_e, Wb + 2 * kHH, bb2, fbuf, kKN, kH, kH, kH, kH, kH, 1.f, 0);

  // ---- trunk MLP over test points: c ----
  gemm_wmma_k<false, 0, true ><<<gQT, 256, 0, stream>>>(tp_e, Wb + 3 * kHH, tb0, cbuf, kQT, kH, kH, kH, kH, kH, 1.f, 0);
  gemm_wmma_k<false, 0, true ><<<gQT, 256, 0, stream>>>(cbuf, Wb + 4 * kHH, tb1, tp_e, kQT, kH, kH, kH, kH, kH, 1.f, 0);
  gemm_wmma_k<false, 0, false><<<gQT, 256, 0, stream>>>(tp_e, Wb + 5 * kHH, tb2, cbuf, kQT, kH, kH, kH, kH, kH, 1.f, 0);

  // ---- q / k / v projections (v stored transposed per batch: [H, NMAX]) ----
  gemm_wmma_k<false, 0, false><<<gQT, 256, 0, stream>>>(cbuf, Wb + 6 * kHH, bq, qbuf,  kQT, kH, kH, kH, kH, kH, 1.f, 0);
  gemm_wmma_k<false, 0, false><<<gKN, 256, 0, stream>>>(fbuf, Wb + 7 * kHH, bk, kbuf,  kKN, kH, kH, kH, kH, kH, 1.f, 0);
  gemm_wmma_k<false, 1, false><<<gKN, 256, 0, stream>>>(fbuf, Wb + 8 * kHH, bv, vTbuf, kKN, kH, kH, kH, kH, kH, 1.f, kN);

  // ---- per-batch masked attention ----
  const float inv_sqrt_h = 0.051031036307982884f;  // 1/sqrt(384)
  const dim3 gSc(kN / 128, kQ / 128);   // scores [1024 x 2048]
  const dim3 gPV(kH / 128, kQ / 128);   // attn   [1024 x 384]
  for (int b = 0; b < kB; ++b) {
    gemm_wmma_k<false, 2, false><<<gSc, 256, 0, stream>>>(
        qbuf + (size_t)b * kQ * kH, kbuf + (size_t)b * kN * kH, nullptr, sc,
        kQ, kN, kH, kH, kH, kN, inv_sqrt_h, 0);
    softmax_k<<<kQ, 256, 0, stream>>>(sc, num_nodes, b);
    gemm_wmma_k<true, 0, false><<<gPV, 256, 0, stream>>>(
        sc, vTbuf + (size_t)b * kH * kN, nullptr, attn + (size_t)b * kQ * kH,
        kQ, kH, kN, kN, kN, kH, 1.f, 0);
  }

  // ---- output projection + MLP head ----
  gemm_wmma_k<false, 0, false><<<gQT, 256, 0, stream>>>(attn, Wb + 9 * kHH,  out_b, t1, kQT, kH, kH, kH, kH, kH, 1.f, 0);
  gemm_wmma_k<false, 0, true ><<<gQT, 256, 0, stream>>>(t1,   Wb + 10 * kHH, ob0,   t2, kQT, kH, kH, kH, kH, kH, 1.f, 0);
  gemm_wmma_k<false, 0, true ><<<gQT, 256, 0, stream>>>(t2,   Wb + 11 * kHH, ob1,   t1, kQT, kH, kH, kH, kH, kH, 1.f, 0);

  head_k<<<(kQT * kOut + 255) / 256, 256, 0, stream>>>(t1, ow2, (float*)d_out, kQT);
}
